// SphericalSparseOperator_90228672954931
// MI455X (gfx1250) — matrix operand order repaired and checked by reading, not compiled, not run on hardware
//
#include <hip/hip_runtime.h>
#include <math.h>

// Problem sizes (fixed by the reference)
#define LNUM 2
#define BB   4
#define NN   2048
#define DD   1024
#define HH   16
#define DHH  64
#define HIDD 4096
#define MM   (BB*NN)      // 8192 tokens
#define LN_EPS 1e-5f

typedef __attribute__((ext_vector_type(16))) __bf16 v16bf;
typedef __attribute__((ext_vector_type(8)))  __bf16 v8bf;
typedef __attribute__((ext_vector_type(8)))  float  v8f;

// gfx1250 async global->LDS path (guarded; falls back to load + ds_store)
#if defined(__gfx1250__) && \
    __has_builtin(__builtin_amdgcn_global_load_async_to_lds_b128) && \
    __has_builtin(__builtin_amdgcn_s_wait_asynccnt)
#define USE_ASYNC_LDS 1
#else
#define USE_ASYNC_LDS 0
#endif

#if USE_ASYNC_LDS
// Builtin signature (from hipcc diagnostic): param0 = int4 addrspace(1)*,
// param1 = int4 addrspace(3)*, then two integer immediates (offset, cpol).
typedef int v4i_gcc __attribute__((vector_size(16)));
typedef __attribute__((address_space(1))) v4i_gcc* as1_v4i;
typedef __attribute__((address_space(3))) v4i_gcc* as3_v4i;

static __device__ inline void async_g2l_b128(const __bf16* g, __bf16* l) {
  // generic LDS pointer low 32 bits == LDS offset (shared aperture rule)
  __builtin_amdgcn_global_load_async_to_lds_b128(
      (as1_v4i)(uintptr_t)g,
      (as3_v4i)(uint32_t)(uintptr_t)l,
      0, 0);
}
#endif

static __device__ inline void copy_g2l_b128(const __bf16* g, __bf16* l) {
#if USE_ASYNC_LDS
  async_g2l_b128(g, l);
#else
  *(v8bf*)l = *(const v8bf*)g;   // global_load_b128 + ds_store_b128
#endif
}

// ---------------------------------------------------------------------------
// WMMA fragment loaders (CDNA5 16-bit layouts, CDNA5 ISA §7.12.2)
// A (16x32): lane L(0-15) row M=L, elems 0..7 -> K=hi*8+0..7, 8..15 -> K=16+hi*8+..
// B (32x16): lane L col N=L&15, elems 0..7 -> K=hi*8+0..7, 8..15 -> K=16+hi*8+..
// ---------------------------------------------------------------------------
static __device__ inline v16bf load_frag_a(const __bf16* src, int ld) {
  int lane = threadIdx.x & 31;
  int row = lane & 15, hi = lane >> 4;
  const __bf16* p = src + (size_t)row * ld + hi * 8;
  v8bf x0 = *(const v8bf*)p;
  v8bf x1 = *(const v8bf*)(p + 16);
  v16bf r;
#pragma unroll
  for (int i = 0; i < 8; ++i) { r[i] = x0[i]; r[i + 8] = x1[i]; }
  return r;
}

static __device__ inline v16bf load_frag_b(const __bf16* src, int ld) {
  int lane = threadIdx.x & 31;
  int col = lane & 15, hi = lane >> 4;
  const __bf16* p = src + (size_t)col * ld + hi * 8;
  v8bf x0 = *(const v8bf*)p;
  v8bf x1 = *(const v8bf*)(p + 16);
  v16bf r;
#pragma unroll
  for (int i = 0; i < 8; ++i) { r[i] = x0[i]; r[i + 8] = x1[i]; }
  return r;
}

static __device__ inline v8f wmma_bf16(v16bf a, v16bf b, v8f c) {
  return __builtin_amdgcn_wmma_f32_16x16x32_bf16(
      /*neg_a=*/false, a, /*neg_b=*/false, b,
      /*c_mod=*/(short)0, c, /*reuse_a=*/false, /*reuse_b=*/false);
}

// ---------------------------------------------------------------------------
// Generic GEMM: C[M,N] = A[M,K] (bf16, row-major) x W[N,K]^T (bf16, row-major)
// Block = 8 waves, block tile 128x128, wave tile 64x32 (4x2 WMMA tiles).
// A/B tiles double-buffered through LDS; fill uses GLOBAL_LOAD_ASYNC_TO_LDS
// (ASYNCcnt) when available.
// EPI: 0 = +bias, store f32   1 = +bias +residual, store f32
//      2 = +bias, exact GELU, store bf16
// ---------------------------------------------------------------------------
template <int EPI>
__global__ __launch_bounds__(256) void gemm_bf16_k(
    const __bf16* __restrict__ A, const __bf16* __restrict__ W,
    const float* __restrict__ bias, float* __restrict__ Cf,
    __bf16* __restrict__ Cb, const float* __restrict__ res,
    int M, int N, int K) {
  __shared__ __align__(16) __bf16 lA[2][128 * 32];
  __shared__ __align__(16) __bf16 lB[2][128 * 32];

  int t = threadIdx.x;
  int wave = t >> 5;
  int lane = t & 31;
  int lm = lane & 15, hi = lane >> 4;
  int wm = wave & 1, wn = wave >> 1;
  int m0 = wm * 64;                 // block-relative
  int n0 = wn * 32;
  int mBase = blockIdx.y * 128;
  int nBase = blockIdx.x * 128;

  const __bf16* gA = A + (size_t)mBase * K;
  const __bf16* gW = W + (size_t)nBase * K;

  // flat indices this thread stages: 2 chunks of 8 bf16 per tile per matrix
  int f0 = t * 8;
  int f1 = t * 8 + 2048;

  auto issue = [&](int buf, int k0) {
    int r0 = f0 >> 5, c0 = f0 & 31;
    int r1 = f1 >> 5, c1 = f1 & 31;
    copy_g2l_b128(gA + (size_t)r0 * K + k0 + c0, &lA[buf][f0]);
    copy_g2l_b128(gA + (size_t)r1 * K + k0 + c1, &lA[buf][f1]);
    copy_g2l_b128(gW + (size_t)r0 * K + k0 + c0, &lB[buf][f0]);
    copy_g2l_b128(gW + (size_t)r1 * K + k0 + c1, &lB[buf][f1]);
  };

  v8f acc[4][2];
#pragma unroll
  for (int i = 0; i < 4; ++i)
#pragma unroll
    for (int j = 0; j < 2; ++j)
#pragma unroll
      for (int r = 0; r < 8; ++r) acc[i][j][r] = 0.f;

  int nIter = K >> 5;
  issue(0, 0);
  for (int kb = 0; kb < nIter; ++kb) {
    int buf = kb & 1;
    if (kb + 1 < nIter) {
      issue(buf ^ 1, (kb + 1) << 5);
#if USE_ASYNC_LDS
      __builtin_amdgcn_s_wait_asynccnt(4);   // previous tile landed
#endif
    } else {
#if USE_ASYNC_LDS
      __builtin_amdgcn_s_wait_asynccnt(0);
#endif
    }
    __syncthreads();

    v16bf a[4], b[2];
#pragma unroll
    for (int i = 0; i < 4; ++i)
      a[i] = load_frag_a(&lA[buf][(m0 + 16 * i) * 32], 32);
#pragma unroll
    for (int j = 0; j < 2; ++j)
      b[j] = load_frag_b(&lB[buf][(n0 + 16 * j) * 32], 32);
#pragma unroll
    for (int i = 0; i < 4; ++i)
#pragma unroll
      for (int j = 0; j < 2; ++j)
        acc[i][j] = wmma_bf16(a[i], b[j], acc[i][j]);

    __syncthreads();   // buffer 'buf' may be overwritten next iteration
  }

#pragma unroll
  for (int i = 0; i < 4; ++i) {
#pragma unroll
    for (int j = 0; j < 2; ++j) {
      int ng = nBase + n0 + 16 * j + lm;
      float bv = bias[ng];
#pragma unroll
      for (int r = 0; r < 8; ++r) {
        int mg = mBase + m0 + 16 * i + 8 * hi + r;
        size_t idx = (size_t)mg * N + ng;
        float v = acc[i][j][r] + bv;
        if (EPI == 0) {
          Cf[idx] = v;
        } else if (EPI == 1) {
          Cf[idx] = v + res[idx];
        } else {  // exact GELU (erf), bf16 output
          float g = 0.5f * v * (1.0f + erff(v * 0.70710678118654752f));
          Cb[idx] = (__bf16)g;
        }
      }
    }
  }
}

// ---------------------------------------------------------------------------
// Flash attention: one wave per (b, h, 16-query tile). Online softmax,
// WMMA for Q*K^T and P*V. P transits LDS for C-layout -> A-layout transpose.
// ---------------------------------------------------------------------------
__global__ __launch_bounds__(32) void flash_attn_k(
    const __bf16* __restrict__ Q, const __bf16* __restrict__ Kc,
    const __bf16* __restrict__ Vt, __bf16* __restrict__ O) {
  __shared__ __align__(16) __bf16 ptile[16 * 32];
  int lane = threadIdx.x & 31;
  int lm = lane & 15, hi = lane >> 4;
  int qi = blockIdx.x, h = blockIdx.y, b = blockIdx.z;

  const __bf16* qbase = Q + ((size_t)b * NN + qi * 16) * DD + h * DHH;
  v16bf qf0 = load_frag_a(qbase, DD);
  v16bf qf1 = load_frag_a(qbase + 32, DD);

  float mrun[8], lrun[8];
#pragma unroll
  for (int r = 0; r < 8; ++r) { mrun[r] = -1e30f; lrun[r] = 0.f; }
  v8f o[4];
#pragma unroll
  for (int c = 0; c < 4; ++c)
#pragma unroll
    for (int r = 0; r < 8; ++r) o[c][r] = 0.f;

  for (int kb = 0; kb < NN / 32; ++kb) {
    // S = Q * K^T for a 16x32 key block (two 16x16 C tiles)
    v8f s0, s1;
#pragma unroll
    for (int r = 0; r < 8; ++r) { s0[r] = 0.f; s1[r] = 0.f; }
    const __bf16* k0p = Kc + ((size_t)b * NN + kb * 32) * DD + h * DHH;
    const __bf16* k1p = k0p + (size_t)16 * DD;
    s0 = wmma_bf16(qf0, load_frag_b(k0p, DD), s0);
    s0 = wmma_bf16(qf1, load_frag_b(k0p + 32, DD), s0);
    s1 = wmma_bf16(qf0, load_frag_b(k1p, DD), s1);
    s1 = wmma_bf16(qf1, load_frag_b(k1p + 32, DD), s1);

    float alpha[8];
#pragma unroll
    for (int r = 0; r < 8; ++r) {
      float a0 = s0[r] * 0.125f;          // 1/sqrt(64)
      float a1 = s1[r] * 0.125f;
      float mx = fmaxf(a0, a1);
#pragma unroll
      for (int mk = 1; mk < 16; mk <<= 1) mx = fmaxf(mx, __shfl_xor(mx, mk, 32));
      float mnew = fmaxf(mrun[r], mx);
      float p0 = __expf(a0 - mnew), p1 = __expf(a1 - mnew);
      float sum = p0 + p1;
#pragma unroll
      for (int mk = 1; mk < 16; mk <<= 1) sum += __shfl_xor(sum, mk, 32);
      alpha[r] = __expf(mrun[r] - mnew);
      lrun[r] = lrun[r] * alpha[r] + sum;
      mrun[r] = mnew;
      // C layout -> LDS row-major [query 16][key 32]
      ptile[(r + 8 * hi) * 32 + lm] = (__bf16)p0;
      ptile[(r + 8 * hi) * 32 + 16 + lm] = (__bf16)p1;
    }
#pragma unroll
    for (int c = 0; c < 4; ++c)
#pragma unroll
      for (int r = 0; r < 8; ++r) o[c][r] *= alpha[r];
    __syncthreads();
    v16bf pa = load_frag_a(ptile, 32);
    __syncthreads();
#pragma unroll
    for (int c = 0; c < 4; ++c) {
      const __bf16* vp = Vt + (((size_t)(b * HH + h)) * DHH + c * 16) * NN + kb * 32;
      o[c] = wmma_bf16(pa, load_frag_b(vp, NN), o[c]);
    }
  }

#pragma unroll
  for (int r = 0; r < 8; ++r) {
    float inv = 1.0f / lrun[r];
    size_t rowg = (size_t)b * NN + (size_t)qi * 16 + 8 * hi + r;
#pragma unroll
    for (int c = 0; c < 4; ++c)
      O[rowg * DD + h * DHH + c * 16 + lm] = (__bf16)(o[c][r] * inv);
  }
}

// ---------------------------------------------------------------------------
// LayerNorm: one wave per row of D=1024; writes fp32 (residual base) + bf16.
// ---------------------------------------------------------------------------
__global__ __launch_bounds__(256) void layernorm_k(
    const float* __restrict__ x, const float* __restrict__ g,
    const float* __restrict__ bta, float* __restrict__ of,
    __bf16* __restrict__ ob) {
  int wave = threadIdx.x >> 5, lane = threadIdx.x & 31;
  size_t row = (size_t)blockIdx.x * 8 + wave;
  const float* xr = x + row * DD;
  float s = 0.f, sq = 0.f;
  for (int d = lane; d < DD; d += 32) { float v = xr[d]; s += v; sq += v * v; }
#pragma unroll
  for (int mk = 1; mk < 32; mk <<= 1) {
    s += __shfl_xor(s, mk, 32);
    sq += __shfl_xor(sq, mk, 32);
  }
  float mean = s * (1.f / DD);
  float var = sq * (1.f / DD) - mean * mean;
  float rstd = rsqrtf(var + LN_EPS);
  for (int d = lane; d < DD; d += 32) {
    float v = (xr[d] - mean) * rstd * g[d] + bta[d];
    of[row * DD + d] = v;
    ob[row * DD + d] = (__bf16)v;
  }
}

// fp32 -> bf16 weight conversion
__global__ __launch_bounds__(256) void cvt_bf16_k(const float* __restrict__ in,
                                                  __bf16* __restrict__ out,
                                                  int n) {
  int i = blockIdx.x * 256 + threadIdx.x;
  if (i < n) out[i] = (__bf16)in[i];
}

// Split QKV (fp32, bias already applied) into q/k bf16 [B,N,D] and
// per-head-transposed V bf16 [B,H,DH,N].
__global__ __launch_bounds__(256) void split_qkv_k(
    const float* __restrict__ qkv, __bf16* __restrict__ qb,
    __bf16* __restrict__ kb, __bf16* __restrict__ vt) {
  int idx = blockIdx.x * 256 + threadIdx.x;  // over B*N*D
  int b = idx / (NN * DD);
  int rem = idx - b * (NN * DD);
  int n = rem / DD;
  int d = rem - n * DD;
  const float* row = qkv + ((size_t)b * NN + n) * (3 * DD);
  qb[idx] = (__bf16)row[d];
  kb[idx] = (__bf16)row[DD + d];
  int h = d >> 6, dh = d & 63;
  vt[(((size_t)b * HH + h) * DHH + dh) * NN + n] = (__bf16)row[2 * DD + d];
}

// ---------------------------------------------------------------------------
extern "C" void kernel_launch(void* const* d_in, const int* in_sizes, int n_in,
                              void* d_out, int out_size, void* d_ws,
                              size_t ws_size, hipStream_t stream) {
  const float* tokens = (const float*)d_in[0];
  const float* ln1_g  = (const float*)d_in[1];
  const float* ln1_b  = (const float*)d_in[2];
  const float* qkv_w  = (const float*)d_in[3];
  const float* qkv_b  = (const float*)d_in[4];
  const float* out_w  = (const float*)d_in[5];
  const float* out_b  = (const float*)d_in[6];
  const float* ln2_g  = (const float*)d_in[7];
  const float* ln2_b  = (const float*)d_in[8];
  const float* fc1_w  = (const float*)d_in[9];
  const float* fc1_b  = (const float*)d_in[10];
  const float* fc2_w  = (const float*)d_in[11];
  const float* fc2_b  = (const float*)d_in[12];
  float* out = (float*)d_out;

  const size_t BND = (size_t)MM * DD;  // 8,388,608
  char* base = (char*)d_ws;
  size_t off = 0;
  auto alloc = [&](size_t bytes) -> void* {
    off = (off + 255) & ~(size_t)255;
    void* p = base + off;
    off += bytes;
    return p;
  };
  float*  xf    = (float*)alloc(BND * 4);           // residual stream
  float*  xn    = (float*)alloc(BND * 4);           // normed (fp32, residual base)
  float*  qkvf  = (float*)alloc(3 * BND * 4);       // qkv fp32
  __bf16* xnb   = (__bf16*)alloc(BND * 2);          // normed bf16
  __bf16* qb    = (__bf16*)alloc(BND * 2);
  __bf16* kb    = (__bf16*)alloc(BND * 2);
  __bf16* vtb   = (__bf16*)alloc(BND * 2);          // [B,H,DH,N]
  __bf16* attnb = (__bf16*)alloc(BND * 2);
  __bf16* hb    = (__bf16*)alloc(4 * BND * 2);      // FC1 activations
  __bf16* wb    = (__bf16*)alloc((size_t)HIDD * DD * 2);  // bf16 weight scratch

  dim3 blk256(256);
  dim3 gLN(MM / 8);
  dim3 gSplit((int)(BND / 256));
  dim3 gFlash(NN / 16, HH, BB);

  for (int i = 0; i < LNUM; ++i) {
    const float* xin = (i == 0) ? tokens : xf;

    // LN1
    hipLaunchKernelGGL(layernorm_k, gLN, blk256, 0, stream, xin,
                       ln1_g + (size_t)i * DD, ln1_b + (size_t)i * DD, xn, xnb);
    // QKV = xn @ qkv_w^T + qkv_b   [M, 3D]
    hipLaunchKernelGGL(cvt_bf16_k, dim3((3 * DD * DD + 255) / 256), blk256, 0,
                       stream, qkv_w + (size_t)i * 3 * DD * DD, wb, 3 * DD * DD);
    hipLaunchKernelGGL((gemm_bf16_k<0>), dim3(3 * DD / 128, MM / 128), blk256, 0,
                       stream, xnb, wb, qkv_b + (size_t)i * 3 * DD, qkvf,
                       (__bf16*)nullptr, (const float*)nullptr, MM, 3 * DD, DD);
    hipLaunchKernelGGL(split_qkv_k, gSplit, blk256, 0, stream, qkvf, qb, kb, vtb);
    // attention
    hipLaunchKernelGGL(flash_attn_k, gFlash, dim3(32), 0, stream, qb, kb, vtb,
                       attnb);
    // out-proj + residual on normed tokens (faithful to reference)
    hipLaunchKernelGGL(cvt_bf16_k, dim3((DD * DD + 255) / 256), blk256, 0,
                       stream, out_w + (size_t)i * DD * DD, wb, DD * DD);
    hipLaunchKernelGGL((gemm_bf16_k<1>), dim3(DD / 128, MM / 128), blk256, 0,
                       stream, attnb, wb, out_b + (size_t)i * DD, xf,
                       (__bf16*)nullptr, xn, MM, DD, DD);
    // LN2
    hipLaunchKernelGGL(layernorm_k, gLN, blk256, 0, stream, xf,
                       ln2_g + (size_t)i * DD, ln2_b + (size_t)i * DD, xn, xnb);
    // FC1 + exact GELU -> bf16
    hipLaunchKernelGGL(cvt_bf16_k, dim3((HIDD * DD + 255) / 256), blk256, 0,
                       stream, fc1_w + (size_t)i * HIDD * DD, wb, HIDD * DD);
    hipLaunchKernelGGL((gemm_bf16_k<2>), dim3(HIDD / 128, MM / 128), blk256, 0,
                       stream, xnb, wb, fc1_b + (size_t)i * HIDD,
                       (float*)nullptr, hb, (const float*)nullptr, MM, HIDD, DD);
    // FC2 + residual; last layer writes d_out directly
    float* dst = (i == LNUM - 1) ? out : xf;
    hipLaunchKernelGGL(cvt_bf16_k, dim3((DD * HIDD + 255) / 256), blk256, 0,
                       stream, fc2_w + (size_t)i * DD * HIDD, wb, DD * HIDD);
    hipLaunchKernelGGL((gemm_bf16_k<1>), dim3(DD / 128, MM / 128), blk256, 0,
                       stream, hb, wb, fc2_b + (size_t)i * DD, dst,
                       (__bf16*)nullptr, xn, MM, DD, HIDD);
  }
}
